// HMM_11596411699492
// MI455X (gfx1250) — matrix operand order repaired and checked by reading, not compiled, not run on hardware
//
#include <hip/hip_runtime.h>
#include <hip/hip_bf16.h>
#include <math.h>

// ---------------- problem constants ----------------
constexpr int S    = 2048;     // states
constexpr int NOBS = 32000;    // observations
constexpr int T    = 8192;     // time steps
constexpr int NWG  = 64;       // persistent workgroups
constexpr int ROWS = S / NWG;  // 32 rows of A per workgroup
constexpr int BLK  = 256;      // 8 wave32s
constexpr int LDA  = S + 8;    // padded bf16 row stride (4112B -> conflict-free b128)

typedef __attribute__((ext_vector_type(16))) __bf16 v16bf;
typedef __attribute__((ext_vector_type(8)))  __bf16 bf16x8;
typedef __attribute__((ext_vector_type(8)))  float  v8f;

// ---------------- small prep kernels ----------------
__global__ void k_zero_bar(int* bar) {
    bar[blockIdx.x * BLK + threadIdx.x] = 0;
}

__global__ void k_erowsum(const float* __restrict__ E, float* __restrict__ erow) {
    __shared__ float red[BLK];
    const int row = blockIdx.x;
    float s = 0.f;
    for (int j = threadIdx.x; j < NOBS; j += BLK) s += E[(size_t)row * NOBS + j];
    red[threadIdx.x] = s; __syncthreads();
    for (int off = BLK / 2; off >= 1; off >>= 1) {
        if ((int)threadIdx.x < off) red[threadIdx.x] += red[threadIdx.x + off];
        __syncthreads();
    }
    if (threadIdx.x == 0) erow[row] = red[0];
}

__global__ void k_spsum(const float* __restrict__ sp, float* __restrict__ spsum) {
    __shared__ float red[BLK];
    float s = 0.f;
    for (int j = threadIdx.x; j < S; j += BLK) s += sp[j];
    red[threadIdx.x] = s; __syncthreads();
    for (int off = BLK / 2; off >= 1; off >>= 1) {
        if ((int)threadIdx.x < off) red[threadIdx.x] += red[threadIdx.x + off];
        __syncthreads();
    }
    if (threadIdx.x == 0) spsum[0] = red[0];
}

// p_hat_0[i] = (sp[i]/sum sp) * Enorm[i, obs[0]]
__global__ void k_p0(const float* __restrict__ sp, const float* __restrict__ E,
                     const int* __restrict__ obs, const float* __restrict__ erow,
                     const float* __restrict__ spsum, float* __restrict__ pbuf) {
    const int i = blockIdx.x * BLK + threadIdx.x;
    const int o = obs[0];
    pbuf[i] = (sp[i] / spsum[0]) * (E[(size_t)i * NOBS + o] / erow[i]);
}

// ---------------- main persistent forward kernel ----------------
__global__ void __launch_bounds__(BLK)
hmm_forward(const float* __restrict__ Atr, const float* __restrict__ E,
            const int* __restrict__ obs, const float* __restrict__ erow,
            float* __restrict__ pbuf /*[2][S]*/, int* __restrict__ bar /*[T]*/,
            float* __restrict__ out /*[S][T]*/) {
    __shared__ __attribute__((aligned(32))) __bf16 Ald[ROWS * LDA]; // 128.5 KB bf16 A slab
    __shared__ __attribute__((aligned(32))) __bf16 qld[S];          // replicated B vector
    __shared__ float ptmp[S];          // fp32 staging of p_prev
    __shared__ float sred[BLK];        // sum reduction
    __shared__ float vbuf[8][16];      // per-wave partial GEMV results
    __shared__ float abuf[32][33];     // 32 rows x 32 time-steps alpha staging

    const int tid  = threadIdx.x;
    const int lane = tid & 31;
    const int wv   = tid >> 5;          // 0..7
    const int rowbase = blockIdx.x * ROWS;

    // ---- load + row-normalize A slab into LDS as bf16 (once) ----
    for (int rr = 0; rr < 4; ++rr) {
        const int r = wv + rr * 8;
        const float* arow = Atr + (size_t)(rowbase + r) * S;
        float s = 0.f;
        for (int j = lane; j < S; j += 32) s += arow[j];
        for (int msk = 16; msk >= 1; msk >>= 1) s += __shfl_xor(s, msk, 32);
        const float inv = 1.0f / s;
        for (int j = lane; j < S; j += 32)
            Ald[r * LDA + j] = (__bf16)(arow[j] * inv);
    }

    // per-thread emission row reciprocal and alpha[.,0]
    float einv = 0.f;
    int   irow = rowbase;
    if (tid < 32) {
        irow = rowbase + tid;
        einv = 1.0f / erow[irow];
        abuf[tid][0] = logf(pbuf[irow]);   // alpha_0 = log(p_hat_0), c_0 = 0
    }
    __syncthreads();

    const int tile = wv >> 2;      // which 16-row tile (0/1)
    const int kq   = wv & 3;       // K quarter (512 wide)
    const int h    = lane >> 4;    // fragment half
    const int m    = lane & 15;    // row within tile

    float c_acc = 0.f;             // running sum of log(scale)

    for (int t = 1; t < T; ++t) {
        const int prev = (t - 1) & 1;

        // early-issue the emission gather for this step (HBM latency hidden)
        float eval = 0.f;
        if (tid < 32) {
            const int o = obs[t];
            eval = E[(size_t)irow * NOBS + o];
        }

        // ---- stage p_prev into LDS + deterministic sum ----
        const float* pp = pbuf + prev * S;
        float ls = 0.f;
        for (int j = tid; j < S; j += BLK) { const float v = pp[j]; ptmp[j] = v; ls += v; }
        sred[tid] = ls;
        __syncthreads();
        for (int off = BLK / 2; off >= 1; off >>= 1) {
            if (tid < off) sred[tid] += sred[tid + off];
            __syncthreads();
        }
        const float s_prev = sred[0];
        const float inv_s  = 1.0f / s_prev;
        c_acc += logf(s_prev);

        // ---- build normalized bf16 q vector ----
        for (int j = tid; j < S; j += BLK) qld[j] = (__bf16)(ptmp[j] * inv_s);
        __syncthreads();

        // ---- GEMV tile via v_wmma_f32_16x16x32_bf16 (B = q replicated in 16 cols) ----
        v8f acc = {};
        const __bf16* arow0 = &Ald[(tile * 16 + m) * LDA];
        const int kbase = kq * 512;
        #pragma unroll
        for (int cch = 0; cch < 16; ++cch) {
            const int k0 = kbase + cch * 32;
            const bf16x8 lo = *(const bf16x8*)(arow0 + k0 + h * 16);
            const bf16x8 hi = *(const bf16x8*)(arow0 + k0 + h * 16 + 8);
            const v16bf a = __builtin_shufflevector(lo, hi,
                0, 1, 2, 3, 4, 5, 6, 7, 8, 9, 10, 11, 12, 13, 14, 15);
            const v16bf b = *(const v16bf*)(&qld[k0 + h * 16]);
            acc = __builtin_amdgcn_wmma_f32_16x16x32_bf16(
                false, a, false, b, (short)0, acc, false, false);
        }
        // columns are replicated; lanes holding N==0 extract the vector
        if (m == 0) {
            #pragma unroll
            for (int r = 0; r < 8; ++r) vbuf[wv][h * 8 + r] = acc[r];
        }
        __syncthreads();

        // ---- combine K-partials, apply emission, write state + alpha ----
        if (tid < 32) {
            const int tl = tid >> 4, mm = tid & 15;
            const float v = vbuf[tl * 4 + 0][mm] + vbuf[tl * 4 + 1][mm]
                          + vbuf[tl * 4 + 2][mm] + vbuf[tl * 4 + 3][mm];
            const float ph = v * (eval * einv);
            pbuf[(t & 1) * S + irow] = ph;
            abuf[tid][t & 31] = logf(ph) + c_acc;
            if (t + 1 < T) {   // prefetch next emission column
                const int on = obs[t + 1];
                __builtin_prefetch(&E[(size_t)irow * NOBS + on], 0, 0);
            }
        }

        // ---- coalesced alpha flush every 32 steps ----
        if ((t & 31) == 31) {
            __syncthreads();
            const int t0 = t - 31;
            for (int e = tid; e < 32 * 32; e += BLK) {
                const int r = e >> 5, cc = e & 31;
                out[(size_t)(rowbase + r) * T + (t0 + cc)] = abuf[r][cc];
            }
        }

        // ---- device-wide barrier (monotonic per-step counter) ----
        __threadfence();
        __syncthreads();
        if (tid == 0) {
            __hip_atomic_fetch_add(&bar[t], 1, __ATOMIC_RELEASE, __HIP_MEMORY_SCOPE_AGENT);
            while (__hip_atomic_load(&bar[t], __ATOMIC_ACQUIRE, __HIP_MEMORY_SCOPE_AGENT)
                   < (int)gridDim.x) {
                __builtin_amdgcn_s_sleep(2);
            }
        }
        __syncthreads();
    }
}

// ---------------- host launcher ----------------
extern "C" void kernel_launch(void* const* d_in, const int* in_sizes, int n_in,
                              void* d_out, int out_size, void* d_ws, size_t ws_size,
                              hipStream_t stream) {
    const float* sp  = (const float*)d_in[0];   // (S,)
    const float* A   = (const float*)d_in[1];   // (S,S)
    const float* E   = (const float*)d_in[2];   // (S,NOBS)
    const int*   obs = (const int*)d_in[3];     // (T,)
    float* out = (float*)d_out;                 // (S,T)

    char* ws = (char*)d_ws;
    int*   bar   = (int*)ws;                                    // T ints   (32 KB)
    float* pbuf  = (float*)(ws + (size_t)T * 4);                // 2*S fp32 (16 KB)
    float* erow  = (float*)(ws + (size_t)T * 4 + 2 * S * 4);    // S fp32   (8 KB)
    float* spsum = (float*)(ws + (size_t)T * 4 + 3 * S * 4);    // 1 fp32

    k_zero_bar<<<T / BLK, BLK, 0, stream>>>(bar);
    k_erowsum<<<S, BLK, 0, stream>>>(E, erow);
    k_spsum<<<1, BLK, 0, stream>>>(sp, spsum);
    k_p0<<<S / BLK, BLK, 0, stream>>>(sp, E, obs, erow, spsum, pbuf);
    hmm_forward<<<NWG, BLK, 0, stream>>>(A, E, obs, erow, pbuf, bar, out);
}
